// VE_MultiHead_SelfAttention_86552180949169
// MI455X (gfx1250) — matrix operand order repaired and checked by reading, not compiled, not run on hardware
//
#include <hip/hip_runtime.h>
#include <hip/hip_bf16.h>
#include <cstdint>

typedef __attribute__((ext_vector_type(16))) _Float16 v16h;
typedef __attribute__((ext_vector_type(8)))  _Float16 v8h;
typedef __attribute__((ext_vector_type(8)))  float    v8f;
typedef _Float16 half_t;

#define BATCH 128
#define SEQ   196
#define DIM   768
#define HEADS 12
#define HDIM  64
#define ROWS  (BATCH * SEQ)   // 25088
#define QKVD  (3 * DIM)       // 2304
#define SEQ_P16 208           // 13 * 16 (col padding for S tiles)
#define SEQ_P32 224           // 7 * 32  (k padding for O GEMM)
#define NT    13              // S col tiles
#define KT7   7               // O GEMM k tiles
#define AWAVES 8

// ---------------- WMMA fragment helpers (contiguous k order, wave32) -------

// A 16x32 f16: lanes0-15 row M=lane, e0..15 -> K=k0+e
//              lanes16-31 row M=lane-16, K=k0+16+e (contiguous, matches B).
__device__ __forceinline__ v16h load_a_frag(const half_t* base, int ld, int row,
                                            int k0, int lane) {
  const int off = (lane & 16) ? 16 : 0;
  const half_t* p = base + (size_t)row * ld + k0 + off;
  return *(const v16h*)(p);
}

// B 32x16 f16 for C[m,n] = sum_k A[m,k]*W[n,k]: lane%16 selects output col n
// (a row of W), lanes16-31 take the upper 16 of K. 16 contiguous halves.
__device__ __forceinline__ v16h load_b_frag(const half_t* base, int ld, int nrow,
                                            int k0, int lane) {
  const half_t* p = base + (size_t)nrow * ld + k0 + ((lane & 16) ? 16 : 0);
  return *(const v16h*)(p);
}

__device__ __forceinline__ v8f wmma_f16(v16h a, v16h b, v8f c) {
  return __builtin_amdgcn_wmma_f32_16x16x32_f16(false, a, false, b, (short)0, c,
                                                false, false);
}

// ---------------- CDNA5 async global->LDS (ASYNCcnt path) ------------------
// Per-lane copy of 16B: LDS[vdst_lds_addr] = MEM[vaddr64]. Generic pointers to
// __shared__ carry the wave-relative LDS byte address in their low 32 bits
// (flat LDS aperture truncates, ISA 10.2), which is what VDST expects.
__device__ __forceinline__ void async_g2l_b128(void* lds_ptr, const void* gptr) {
  uint32_t lds_off = (uint32_t)(uintptr_t)lds_ptr;
  uint64_t gaddr = (uint64_t)(uintptr_t)gptr;
  asm volatile("global_load_async_to_lds_b128 %0, %1, off"
               :: "v"(lds_off), "v"(gaddr)
               : "memory");
}
__device__ __forceinline__ void wait_asynccnt0() {
  asm volatile("s_wait_asynccnt 0x0" ::: "memory");
}

// ---------------- Reductions & gates ---------------------------------------

__global__ void k_row_stats(const float* __restrict__ x, half_t* __restrict__ xh,
                            float* __restrict__ maxn, float* __restrict__ aven) {
  __shared__ float smx[256], ssm[256];
  const int r = blockIdx.x, tid = threadIdx.x;
  const float* xr = x + (size_t)r * DIM;
  half_t* xhr = xh + (size_t)r * DIM;
  float mx = -3.0e38f, sm = 0.0f;
  for (int i = tid; i < DIM; i += 256) {
    float v = xr[i];
    xhr[i] = (half_t)v;
    mx = fmaxf(mx, v);
    sm += v;
  }
  smx[tid] = mx; ssm[tid] = sm;
  __syncthreads();
  for (int s = 128; s > 0; s >>= 1) {
    if (tid < s) {
      smx[tid] = fmaxf(smx[tid], smx[tid + s]);
      ssm[tid] += ssm[tid + s];
    }
    __syncthreads();
  }
  if (tid == 0) { maxn[r] = smx[0]; aven[r] = ssm[0] * (1.0f / DIM); }
}

__global__ void k_batch_stats(const float* __restrict__ maxn,
                              const float* __restrict__ aven,
                              float* __restrict__ maxg, float* __restrict__ aveg,
                              float* __restrict__ gm_m, float* __restrict__ gm_s) {
  __shared__ float smx[256], ssm[256];
  __shared__ float s_maxg, s_zmax;
  const int b = blockIdx.x, tid = threadIdx.x;
  float mx = (tid < SEQ) ? maxn[b * SEQ + tid] : -3.0e38f;
  float sm = (tid < SEQ) ? aven[b * SEQ + tid] : 0.0f;
  smx[tid] = mx; ssm[tid] = sm;
  __syncthreads();
  for (int s = 128; s > 0; s >>= 1) {
    if (tid < s) {
      smx[tid] = fmaxf(smx[tid], smx[tid + s]);
      ssm[tid] += ssm[tid + s];
    }
    __syncthreads();
  }
  if (tid == 0) {
    s_maxg = smx[0];
    maxg[b] = smx[0];
    aveg[b] = ssm[0] * (1.0f / SEQ);
  }
  __syncthreads();
  // softmax over n of maxg*maxn: max then sum(exp)
  float z = (tid < SEQ) ? s_maxg * maxn[b * SEQ + tid] : -3.0e38f;
  smx[tid] = z;
  __syncthreads();
  for (int s = 128; s > 0; s >>= 1) {
    if (tid < s) smx[tid] = fmaxf(smx[tid], smx[tid + s]);
    __syncthreads();
  }
  if (tid == 0) s_zmax = smx[0];
  __syncthreads();
  float e = (tid < SEQ) ? __expf(z - s_zmax) : 0.0f;
  ssm[tid] = e;
  __syncthreads();
  for (int s = 128; s > 0; s >>= 1) {
    if (tid < s) ssm[tid] += ssm[tid + s];
    __syncthreads();
  }
  if (tid == 0) { gm_m[b] = s_zmax; gm_s[b] = ssm[0]; }
}

__global__ void k_col_stats(const float* __restrict__ aven,
                            const float* __restrict__ aveg,
                            float* __restrict__ ga_m, float* __restrict__ ga_s) {
  __shared__ float s0[128], s1[128];
  __shared__ float s_wm;
  const int n = blockIdx.x, b = threadIdx.x;
  float w = aveg[b] * aven[b * SEQ + n];
  s0[b] = w;
  __syncthreads();
  for (int s = 64; s > 0; s >>= 1) {
    if (b < s) s0[b] = fmaxf(s0[b], s0[b + s]);
    __syncthreads();
  }
  if (b == 0) s_wm = s0[0];
  __syncthreads();
  float e = __expf(w - s_wm);
  s1[b] = e;
  __syncthreads();
  for (int s = 64; s > 0; s >>= 1) {
    if (b < s) s1[b] += s1[b + s];
    __syncthreads();
  }
  if (b == 0) { ga_m[n] = s_wm; ga_s[n] = s1[0]; }
}

__global__ void k_gates(const float* __restrict__ maxn, const float* __restrict__ aven,
                        const float* __restrict__ maxg, const float* __restrict__ aveg,
                        const float* __restrict__ gm_m, const float* __restrict__ gm_s,
                        const float* __restrict__ ga_m, const float* __restrict__ ga_s,
                        float* __restrict__ g) {
  const int r = blockIdx.x * 256 + threadIdx.x;
  if (r >= ROWS) return;
  const int b = r / SEQ, n = r - b * SEQ;
  float gm = __expf(maxg[b] * maxn[r] - gm_m[b]) / gm_s[b];
  float ga = __expf(aveg[b] * aven[r] - ga_m[n]) / ga_s[n];
  g[r] = gm * ga;
}

__global__ void k_wconv(const float* __restrict__ wqkv, const float* __restrict__ wproj,
                        half_t* __restrict__ wqkvh, half_t* __restrict__ wprojh) {
  const int i = blockIdx.x * 256 + threadIdx.x;
  if (i < QKVD * DIM) wqkvh[i] = (half_t)wqkv[i];
  const int j = i - QKVD * DIM;
  if (j >= 0 && j < DIM * DIM) wprojh[j] = (half_t)wproj[j];
}

// ---------------- qkv GEMM: (25088x768) x (768x2304) -----------------------
// Epilogue routes q -> Ah, k -> Kh, v*g + x_Exm -> Vh, all [b,h,n,d] f16.

__launch_bounds__(128, 2)
__global__ void k_qkv(const half_t* __restrict__ xh, const half_t* __restrict__ wh,
                      const float* __restrict__ g, const float* __restrict__ x_exm,
                      half_t* __restrict__ Ah, half_t* __restrict__ Kh,
                      half_t* __restrict__ Vh) {
  const int lane = threadIdx.x & 31, wave = threadIdx.x >> 5;
  const int row0 = blockIdx.x * 16;
  const int c0 = blockIdx.y * 64 + wave * 16;
  const int arow = row0 + (lane & 15);
  const int bcol = c0 + (lane & 15);
  v8f acc = {};
#pragma unroll 4
  for (int k0 = 0; k0 < DIM; k0 += 32) {
    v16h a = load_a_frag(xh, DIM, arow, k0, lane);
    v16h b = load_b_frag(wh, DIM, bcol, k0, lane);
    acc = wmma_f16(a, b, acc);
  }
  const int sec = c0 / DIM;          // 768 % 64 == 0: tile never straddles q/k/v
  const int cc = bcol % DIM;
  const int h = cc >> 6, d = cc & 63;
#pragma unroll
  for (int i = 0; i < 8; ++i) {
    const int r = row0 + i + ((lane & 16) ? 8 : 0);
    const int bi = r / SEQ, n = r - bi * SEQ;
    const size_t hoff = (((size_t)bi * HEADS + h) * SEQ + n) * HDIM + d;
    const float v = acc[i];
    if (sec == 0) {
      Ah[hoff] = (half_t)v;                        // q (v added later)
    } else if (sec == 1) {
      Kh[hoff] = (half_t)v;
    } else {
      Vh[hoff] = (half_t)(v * g[r] + x_exm[(size_t)r * DIM + cc]);
    }
  }
}

__global__ void k_addqv(half_t* __restrict__ Ah, const half_t* __restrict__ Vh) {
  const int i = blockIdx.x * 256 + threadIdx.x;
  if (i < ROWS * DIM) Ah[i] = (half_t)((float)Ah[i] + (float)Vh[i]);
}

// ---------------- Attention: S=(q+v)k^T, P=softmax(S*scale), O=P.v ---------

__launch_bounds__(256, 1)
__global__ void k_attn(const half_t* __restrict__ Ah, const half_t* __restrict__ Kh,
                       const half_t* __restrict__ Vh, half_t* __restrict__ Oh) {
  __shared__ half_t Kl[SEQ_P16 * HDIM];           // 26.0 KB
  __shared__ half_t Vt[HDIM * SEQ_P32];           // 28.0 KB (transposed V)
  __shared__ half_t Pl[AWAVES][16 * SEQ_P32];     // 56.0 KB
  const int tid = threadIdx.x, lane = tid & 31, wave = tid >> 5;
  const int bh = blockIdx.x;
  const int b = bh / HEADS, h = bh % HEADS;
  const size_t base = (size_t)bh * SEQ * HDIM;
  const half_t* Ka = Kh + base;
  const half_t* Va = Vh + base;
  const half_t* Aa = Ah + base;

  // Stage K rows via async global->LDS DMA (16B per lane per op, ASYNCcnt).
  // Valid region: SEQ*HDIM halves = 1568 x 16B chunks.
  for (int i = tid; i < (SEQ * HDIM) / 8; i += 256) {
    const int e = i * 8;
    async_g2l_b128(&Kl[e], Ka + e);
  }
  // Zero-fill pad rows 196..207 with ordinary DS stores.
  {
    const int e = SEQ * HDIM + tid * 4;
    if (e < SEQ_P16 * HDIM) *(uint64_t*)(&Kl[e]) = 0ull;
  }
  // Stage V transposed: Vt[d][m] = V[m][d], zero-pad m >= SEQ
  for (int i = tid; i < SEQ_P32 * HDIM; i += 256) {
    const int n = i >> 6, d = i & 63;
    Vt[d * SEQ_P32 + n] = (n < SEQ) ? Va[(size_t)n * HDIM + d] : (half_t)0.0f;
  }
  wait_asynccnt0();      // this wave's async K-tile transfers complete
  __syncthreads();       // all waves' transfers visible block-wide

  const float scale = 0.125f;  // hd^-0.5
  for (int rt = wave; rt < NT; rt += AWAVES) {
    const int nrow = rt * 16 + (lane & 15);
    const int nclamp = (nrow < SEQ) ? nrow : (SEQ - 1);
    const v16h a0 = load_a_frag(Aa, HDIM, nclamp, 0, lane);
    const v16h a1 = load_a_frag(Aa, HDIM, nclamp, 32, lane);

    v8f s[NT];
#pragma unroll
    for (int mt = 0; mt < NT; ++mt) {
      v8f acc = {};
      v16h b0 = load_b_frag(Kl, HDIM, mt * 16 + (lane & 15), 0, lane);
      v16h b1 = load_b_frag(Kl, HDIM, mt * 16 + (lane & 15), 32, lane);
      acc = wmma_f16(a0, b0, acc);
      acc = wmma_f16(a1, b1, acc);
      s[mt] = acc;
    }

    // Row softmax. C-layout: element i -> row M=i+8*(lane>=16), col=16*mt+lane%16.
    const int colbase = lane & 15;
    float mx[8], sum[8];
#pragma unroll
    for (int i = 0; i < 8; ++i) mx[i] = -3.0e38f;
#pragma unroll
    for (int mt = 0; mt < NT; ++mt) {
      const bool valid = (mt * 16 + colbase) < SEQ;
#pragma unroll
      for (int i = 0; i < 8; ++i) {
        float v = valid ? s[mt][i] : -3.0e38f;
        s[mt][i] = v;
        mx[i] = fmaxf(mx[i], v);
      }
    }
#pragma unroll
    for (int i = 0; i < 8; ++i) {
#pragma unroll
      for (int off = 8; off > 0; off >>= 1)
        mx[i] = fmaxf(mx[i], __shfl_xor(mx[i], off, 16));
      sum[i] = 0.0f;
    }
#pragma unroll
    for (int mt = 0; mt < NT; ++mt) {
#pragma unroll
      for (int i = 0; i < 8; ++i) {
        float p = __expf((s[mt][i] - mx[i]) * scale);  // masked -> exp(-huge)=0
        s[mt][i] = p;
        sum[i] += p;
      }
    }
#pragma unroll
    for (int i = 0; i < 8; ++i) {
#pragma unroll
      for (int off = 8; off > 0; off >>= 1) sum[i] += __shfl_xor(sum[i], off, 16);
      sum[i] = 1.0f / sum[i];
    }

    // Store P (f16) to this wave's LDS tile in row-major [16][224]
    half_t* prow = &Pl[wave][0];
#pragma unroll
    for (int i = 0; i < 8; ++i) {
      const int M = i + ((lane & 16) ? 8 : 0);
#pragma unroll
      for (int mt = 0; mt < NT; ++mt)
        prow[M * SEQ_P32 + mt * 16 + colbase] = (half_t)(s[mt][i] * sum[i]);
      prow[M * SEQ_P32 + SEQ_P16 + colbase] = (half_t)0.0f;  // pad cols 208..223
    }

    // O = P (16x224) x V (224x64), via Vt for contiguous B-fragments
    v8f o[4] = {{}, {}, {}, {}};
#pragma unroll
    for (int kt = 0; kt < KT7; ++kt) {
      v16h pa = load_a_frag(prow, SEQ_P32, (lane & 15), kt * 32, lane);
#pragma unroll
      for (int nt = 0; nt < 4; ++nt) {
        v16h vb = load_b_frag(Vt, SEQ_P32, nt * 16 + (lane & 15), kt * 32, lane);
        o[nt] = wmma_f16(pa, vb, o[nt]);
      }
    }
#pragma unroll
    for (int nt = 0; nt < 4; ++nt) {
      const int d = nt * 16 + (lane & 15);
#pragma unroll
      for (int i = 0; i < 8; ++i) {
        const int n = rt * 16 + i + ((lane & 16) ? 8 : 0);
        if (n < SEQ)
          Oh[((size_t)(b * SEQ + n)) * DIM + h * HDIM + d] = (half_t)o[nt][i];
      }
    }
  }
}

// ---------------- Output projection GEMM + bias -----------------------------

__launch_bounds__(128, 2)
__global__ void k_proj(const half_t* __restrict__ Oh, const half_t* __restrict__ wh,
                       const float* __restrict__ bias, float* __restrict__ out) {
  const int lane = threadIdx.x & 31, wave = threadIdx.x >> 5;
  const int row0 = blockIdx.x * 16;
  const int c0 = blockIdx.y * 64 + wave * 16;
  const int arow = row0 + (lane & 15);
  const int bcol = c0 + (lane & 15);
  v8f acc = {};
#pragma unroll 4
  for (int k0 = 0; k0 < DIM; k0 += 32) {
    v16h a = load_a_frag(Oh, DIM, arow, k0, lane);
    v16h b = load_b_frag(wh, DIM, bcol, k0, lane);
    acc = wmma_f16(a, b, acc);
  }
  const float bb = bias[bcol];
#pragma unroll
  for (int i = 0; i < 8; ++i) {
    const int r = row0 + i + ((lane & 16) ? 8 : 0);
    out[(size_t)r * DIM + bcol] = acc[i] + bb;
  }
}

// ---------------- Launcher ---------------------------------------------------

extern "C" void kernel_launch(void* const* d_in, const int* in_sizes, int n_in,
                              void* d_out, int out_size, void* d_ws, size_t ws_size,
                              hipStream_t stream) {
  (void)in_sizes; (void)n_in; (void)out_size; (void)ws_size;
  const float* x     = (const float*)d_in[0];
  const float* x_exm = (const float*)d_in[1];
  const float* wqkv  = (const float*)d_in[2];
  const float* wproj = (const float*)d_in[3];
  const float* bproj = (const float*)d_in[4];
  float* out = (float*)d_out;

  char* ws = (char*)d_ws;
  size_t cur = 0;
  auto alloc = [&](size_t bytes) -> void* {
    void* p = ws + cur;
    cur = (cur + bytes + 255) & ~(size_t)255;
    return p;
  };
  half_t* xh     = (half_t*)alloc((size_t)ROWS * DIM * 2);
  half_t* wqkvh  = (half_t*)alloc((size_t)QKVD * DIM * 2);
  half_t* wprojh = (half_t*)alloc((size_t)DIM * DIM * 2);
  half_t* Kh     = (half_t*)alloc((size_t)ROWS * DIM * 2);
  half_t* Vh     = (half_t*)alloc((size_t)ROWS * DIM * 2);
  half_t* Ahp    = (half_t*)alloc((size_t)ROWS * DIM * 2);
  half_t* Oh     = (half_t*)alloc((size_t)ROWS * DIM * 2);
  float* maxn = (float*)alloc((size_t)ROWS * 4);
  float* aven = (float*)alloc((size_t)ROWS * 4);
  float* gbuf = (float*)alloc((size_t)ROWS * 4);
  float* maxg = (float*)alloc(512);
  float* aveg = (float*)alloc(512);
  float* gm_m = (float*)alloc(512);
  float* gm_s = (float*)alloc(512);
  float* ga_m = (float*)alloc(1024);
  float* ga_s = (float*)alloc(1024);

  k_row_stats<<<ROWS, 256, 0, stream>>>(x, xh, maxn, aven);
  k_batch_stats<<<BATCH, 256, 0, stream>>>(maxn, aven, maxg, aveg, gm_m, gm_s);
  k_col_stats<<<SEQ, 128, 0, stream>>>(aven, aveg, ga_m, ga_s);
  k_gates<<<(ROWS + 255) / 256, 256, 0, stream>>>(maxn, aven, maxg, aveg,
                                                  gm_m, gm_s, ga_m, ga_s, gbuf);
  k_wconv<<<(QKVD * DIM + DIM * DIM + 255) / 256, 256, 0, stream>>>(wqkv, wproj,
                                                                    wqkvh, wprojh);
  k_qkv<<<dim3(ROWS / 16, QKVD / 64), 128, 0, stream>>>(xh, wqkvh, gbuf, x_exm,
                                                        Ahp, Kh, Vh);
  k_addqv<<<(ROWS * DIM + 255) / 256, 256, 0, stream>>>(Ahp, Vh);
  k_attn<<<BATCH * HEADS, 256, 0, stream>>>(Ahp, Kh, Vh, Oh);
  k_proj<<<dim3(ROWS / 16, DIM / 64), 128, 0, stream>>>(Oh, wprojh, bproj, out);
}
